// Mini_to_Large_46961172414968
// MI455X (gfx1250) — matrix operand order repaired and checked by reading, not compiled, not run on hardware
//
#include <hip/hip_runtime.h>
#include <hip/hip_bf16.h>
#include <stdint.h>

// ---------------- types ----------------
typedef __attribute__((ext_vector_type(16))) __bf16 v16bf;
typedef __attribute__((ext_vector_type(8)))  __bf16 v8bf;
typedef __attribute__((ext_vector_type(8)))  float  v8f;

// ---------------- problem constants ----------------
constexpr int kN    = 500000;   // points
constexpr int kM    = 50000;    // clusters
constexpr int kD    = 128;      // feature dim
constexpr int kC    = 131;      // D+3
constexpr int kKP   = 160;      // K padded to multiple of 32
constexpr int kH1   = 64;       // attention hidden
constexpr int kOH   = 128;      // output MLP hidden
constexpr int kON   = 256;      // output MLP out
constexpr int kCOLS = 132;      // agg row stride (padded)

// weight swizzle sizes
constexpr int kW1A = 5 * 4 * 32 * 16;    // attention W1: [ch][nt][lane][e]
constexpr int kW1O = 5 * 8 * 32 * 16;    // output  W1
constexpr int kW2O = 4 * 16 * 32 * 16;   // output  W2

// per-wave LDS in attn kernel: t[16][160] f32 + att[16] + lbl[16]
constexpr int kWaveFloatsB = 16 * kKP + 16 + 16;       // 2592 floats
constexpr size_t kSmemB = (size_t)kW1A * 2 + (size_t)8 * kWaveFloatsB * 4;
constexpr size_t kSmemC = (size_t)(kW1O + kW2O + 8 * 16 * kOH) * 2;

__device__ __forceinline__ void wave_lds_sync() {
  asm volatile("s_wait_dscnt 0" ::: "memory");
  __builtin_amdgcn_wave_barrier();
}

__device__ __forceinline__ v8f wmma_bf16(v16bf a, v16bf b, v8f c) {
  return __builtin_amdgcn_wmma_f32_16x16x32_bf16(false, a, false, b,
                                                 (short)0, c, false, false);
}

// ---------------- kernel 1: zero agg ----------------
__global__ void zero_f4(float4* __restrict__ p, int n4) {
  int i = blockIdx.x * 256 + threadIdx.x;
  if (i < n4) p[i] = make_float4(0.f, 0.f, 0.f, 0.f);
}

// ---------------- kernel 2: attention MLP + scaled scatter-sum ----------------
__global__ __launch_bounds__(256) void attn_scatter_kernel(
    const float* __restrict__ feat, const float* __restrict__ pts,
    const float* __restrict__ ctr,  const int* __restrict__ lbl,
    const float* __restrict__ aW1,  const float* __restrict__ ab1,
    const float* __restrict__ aW2,  const float* __restrict__ ab2,
    float* __restrict__ agg, int numTiles)
{
  extern __shared__ char smem[];
  __bf16* w1 = (__bf16*)smem;                       // [5][4][32][16] bf16
  const int tid = threadIdx.x, lane = tid & 31, wave = tid >> 5;
  const int ln = lane & 15, half = lane >> 4;

  // Pre-swizzle aW1 -> bf16 B-operand layout: element e of lane l for
  // (ch, nt) holds B[k = ch*32 + (e>>3)*16 + (l>>4)*8 + (e&7)][n = nt*16 + (l&15)].
  for (int i = tid; i < kW1A; i += 256) {
    int e = i & 15, l = (i >> 4) & 31, nt = (i >> 9) & 3, ch = i >> 11;
    int k = ch * 32 + (e >> 3) * 16 + (l >> 4) * 8 + (e & 7);
    int n = nt * 16 + (l & 15);
    w1[i] = (k < kC) ? (__bf16)aW1[k * kH1 + n] : (__bf16)0.0f;
  }
  __syncthreads();

  const int tileIdx = blockIdx.x * 8 + wave;        // wave-uniform
  if (tileIdx >= numTiles) return;

  float* t    = (float*)(smem + (size_t)kW1A * 2) + (size_t)wave * kWaveFloatsB;
  float* attp = t + 16 * kKP;
  int*   lblp = (int*)(attp + 16);

  const int base = tileIdx * 16;

  // Stage features global -> LDS via async-DMA path (ASYNCcnt-tracked,
  // no VGPR round trip): per-lane 16B copy, 16 rows.
  {
    unsigned ldsbase = (unsigned)(uintptr_t)t + (unsigned)lane * 16u;
    unsigned long long gbase =
        (unsigned long long)(uintptr_t)(feat + (size_t)base * kD) +
        (unsigned long long)lane * 16ull;
    #pragma unroll
    for (int r = 0; r < 16; ++r) {
      unsigned           l = ldsbase + (unsigned)(r * kKP * 4);
      unsigned long long g = gbase + (unsigned long long)(r * kD * 4);
      asm volatile("global_load_async_to_lds_b128 %0, %1, off"
                   :: "v"(l), "v"(g) : "memory");
    }
  }
  // rel coords + labels + zero padding (regular DS stores, DScnt-tracked)
  if (lane < 16) {
    const int r = lane;
    const int lb = lbl[base + r];
    lblp[r] = lb;
    t[r * kKP + 128] = ctr[lb * 3 + 0] - pts[(size_t)(base + r) * 3 + 0];
    t[r * kKP + 129] = ctr[lb * 3 + 1] - pts[(size_t)(base + r) * 3 + 1];
    t[r * kKP + 130] = ctr[lb * 3 + 2] - pts[(size_t)(base + r) * 3 + 2];
    #pragma unroll
    for (int c = 131; c < kKP; ++c) t[r * kKP + c] = 0.0f;
  }
  asm volatile("s_wait_asynccnt 0x0" ::: "memory");
  wave_lds_sync();

  // GEMM1: [16x160]bf16 x [160x64]bf16 -> f32, 20 WMMAs
  v8f acc[4];
  #pragma unroll
  for (int nt = 0; nt < 4; ++nt) {
    float b = ab1[nt * 16 + ln];
    #pragma unroll
    for (int j = 0; j < 8; ++j) acc[nt][j] = b;
  }
  #pragma unroll
  for (int ch = 0; ch < 5; ++ch) {
    // A layout (16-bit): lane holds row M=ln; elems 0..7 -> K=k0+half*8+i,
    // elems 8..15 -> K=k0+16+half*8+i
    const float* ap = t + ln * kKP + ch * 32 + half * 8;
    float4 l0 = *(const float4*)(ap);
    float4 l1 = *(const float4*)(ap + 4);
    float4 h0 = *(const float4*)(ap + 16);
    float4 h1 = *(const float4*)(ap + 20);
    v16bf A;
    A[0]=(__bf16)l0.x; A[1]=(__bf16)l0.y; A[2]=(__bf16)l0.z; A[3]=(__bf16)l0.w;
    A[4]=(__bf16)l1.x; A[5]=(__bf16)l1.y; A[6]=(__bf16)l1.z; A[7]=(__bf16)l1.w;
    A[8]=(__bf16)h0.x; A[9]=(__bf16)h0.y; A[10]=(__bf16)h0.z; A[11]=(__bf16)h0.w;
    A[12]=(__bf16)h1.x; A[13]=(__bf16)h1.y; A[14]=(__bf16)h1.z; A[15]=(__bf16)h1.w;
    #pragma unroll
    for (int nt = 0; nt < 4; ++nt) {
      v16bf B = *(const v16bf*)(w1 + ((ch * 4 + nt) * 32 + lane) * 16);
      acc[nt] = wmma_bf16(A, B, acc[nt]);
    }
  }

  // attention head: att[m] = sigmoid(relu(h1[m]) . aW2 + ab2), cross-lane reduce
  float w2v[4];
  #pragma unroll
  for (int nt = 0; nt < 4; ++nt) w2v[nt] = aW2[nt * 16 + ln];
  const float b2 = ab2[0];
  float attl[8];
  #pragma unroll
  for (int j = 0; j < 8; ++j) {
    float s = 0.0f;
    #pragma unroll
    for (int nt = 0; nt < 4; ++nt) s += fmaxf(acc[nt][j], 0.0f) * w2v[nt];
    s += __shfl_xor(s, 8);
    s += __shfl_xor(s, 4);
    s += __shfl_xor(s, 2);
    s += __shfl_xor(s, 1);
    attl[j] = 1.0f / (1.0f + __expf(-(s + b2)));
  }
  if (ln == 0) {
    #pragma unroll
    for (int j = 0; j < 8; ++j) attp[half * 8 + j] = attl[j];
  }
  wave_lds_sync();

  // scatter: agg[label[m]][c] += t[m][c] * att[m]  (f32 HW atomics, L2-resident)
  #pragma unroll 1
  for (int m = 0; m < 16; ++m) {
    const float am = attp[m];
    const int   lb = lblp[m];
    float* dst = agg + (size_t)lb * kCOLS;
    for (int c = lane; c < kC; c += 32) {
      __hip_atomic_fetch_add(dst + c, t[m * kKP + c] * am,
                             __ATOMIC_RELAXED, __HIP_MEMORY_SCOPE_AGENT);
    }
  }
}

// ---------------- kernel 3: output MLP over agg ----------------
__global__ __launch_bounds__(256) void out_mlp_kernel(
    const float* __restrict__ agg, const float* __restrict__ oW1,
    const float* __restrict__ ob1, const float* __restrict__ oW2,
    const float* __restrict__ ob2, float* __restrict__ out, int numTiles)
{
  extern __shared__ char smem[];
  __bf16* w1  = (__bf16*)smem;       // [5][8][32][16]
  __bf16* w2  = w1 + kW1O;           // [4][16][32][16]
  __bf16* hst = w2 + kW2O;           // per-wave [16][128] bf16
  const int tid = threadIdx.x, lane = tid & 31, wave = tid >> 5;
  const int ln = lane & 15, half = lane >> 4;

  for (int i = tid; i < kW1O; i += 256) {
    int e = i & 15, l = (i >> 4) & 31, nt = (i >> 9) & 7, ch = i >> 12;
    int k = ch * 32 + (e >> 3) * 16 + (l >> 4) * 8 + (e & 7);
    int n = nt * 16 + (l & 15);
    w1[i] = (k < kC) ? (__bf16)oW1[k * kOH + n] : (__bf16)0.0f;
  }
  for (int i = tid; i < kW2O; i += 256) {
    int e = i & 15, l = (i >> 4) & 31, nt = (i >> 9) & 15, ch = i >> 13;
    int k = ch * 32 + (e >> 3) * 16 + (l >> 4) * 8 + (e & 7);
    int n = nt * 16 + (l & 15);
    w2[i] = (__bf16)oW2[k * kON + n];
  }
  __syncthreads();

  const int tileIdx = blockIdx.x * 8 + wave;        // wave-uniform
  if (tileIdx >= numTiles) return;

  const float* arow = agg + (size_t)(tileIdx * 16 + ln) * kCOLS;

  // GEMM1: [16x160]bf16 x [160x128]bf16 -> f32, 40 WMMAs
  v8f acc[8];
  #pragma unroll
  for (int nt = 0; nt < 8; ++nt) {
    float b = ob1[nt * 16 + ln];
    #pragma unroll
    for (int j = 0; j < 8; ++j) acc[nt][j] = b;
  }
  #pragma unroll
  for (int ch = 0; ch < 5; ++ch) {
    v16bf A;
    if (ch < 4) {
      const float* ap = arow + ch * 32 + half * 8;
      float4 l0 = *(const float4*)(ap);
      float4 l1 = *(const float4*)(ap + 4);
      float4 h0 = *(const float4*)(ap + 16);
      float4 h1 = *(const float4*)(ap + 20);
      A[0]=(__bf16)l0.x; A[1]=(__bf16)l0.y; A[2]=(__bf16)l0.z; A[3]=(__bf16)l0.w;
      A[4]=(__bf16)l1.x; A[5]=(__bf16)l1.y; A[6]=(__bf16)l1.z; A[7]=(__bf16)l1.w;
      A[8]=(__bf16)h0.x; A[9]=(__bf16)h0.y; A[10]=(__bf16)h0.z; A[11]=(__bf16)h0.w;
      A[12]=(__bf16)h1.x; A[13]=(__bf16)h1.y; A[14]=(__bf16)h1.z; A[15]=(__bf16)h1.w;
    } else {
      // K = 128..159: only 128..130 valid
      #pragma unroll
      for (int i = 0; i < 8; ++i) {
        int k = 128 + half * 8 + i;
        A[i]     = (k < kC) ? (__bf16)arow[k] : (__bf16)0.0f;
        A[8 + i] = (__bf16)0.0f;   // K = 144..159 all padding
      }
    }
    #pragma unroll
    for (int nt = 0; nt < 8; ++nt) {
      v16bf B = *(const v16bf*)(w1 + ((ch * 8 + nt) * 32 + lane) * 16);
      acc[nt] = wmma_bf16(A, B, acc[nt]);
    }
  }

  // relu -> bf16 restage to LDS (row-major [16][128])
  __bf16* hs = hst + wave * (16 * kOH);
  #pragma unroll
  for (int nt = 0; nt < 8; ++nt)
    #pragma unroll
    for (int j = 0; j < 8; ++j)
      hs[(8 * half + j) * kOH + nt * 16 + ln] = (__bf16)fmaxf(acc[nt][j], 0.0f);
  wave_lds_sync();

  // GEMM2: [16x128]bf16 x [128x256]bf16 -> f32, 64 WMMAs
  v8f acc2[16];
  #pragma unroll
  for (int nt = 0; nt < 16; ++nt) {
    float b = ob2[nt * 16 + ln];
    #pragma unroll
    for (int j = 0; j < 8; ++j) acc2[nt][j] = b;
  }
  #pragma unroll
  for (int ch = 0; ch < 4; ++ch) {
    const __bf16* ap = hs + ln * kOH + ch * 32 + half * 8;
    v8bf lo = *(const v8bf*)(ap);
    v8bf hi = *(const v8bf*)(ap + 16);
    v16bf A;
    #pragma unroll
    for (int i = 0; i < 8; ++i) { A[i] = lo[i]; A[8 + i] = hi[i]; }
    #pragma unroll
    for (int nt = 0; nt < 16; ++nt) {
      v16bf B = *(const v16bf*)(w2 + ((ch * 16 + nt) * 32 + lane) * 16);
      acc2[nt] = wmma_bf16(A, B, acc2[nt]);
    }
  }

  float* orow = out + (size_t)(tileIdx * 16) * kON;
  #pragma unroll
  for (int nt = 0; nt < 16; ++nt)
    #pragma unroll
    for (int j = 0; j < 8; ++j)
      orow[(8 * half + j) * kON + nt * 16 + ln] = fmaxf(acc2[nt][j], 0.0f);
}

// ---------------- launch ----------------
extern "C" void kernel_launch(void* const* d_in, const int* in_sizes, int n_in,
                              void* d_out, int out_size, void* d_ws, size_t ws_size,
                              hipStream_t stream) {
  const float* feat = (const float*)d_in[0];
  const float* pts  = (const float*)d_in[1];
  const float* ctr  = (const float*)d_in[2];
  const int*   lbl  = (const int*)  d_in[3];
  const float* aW1  = (const float*)d_in[4];
  const float* ab1  = (const float*)d_in[5];
  const float* aW2  = (const float*)d_in[6];
  const float* ab2  = (const float*)d_in[7];
  const float* oW1  = (const float*)d_in[8];
  const float* ob1  = (const float*)d_in[9];
  const float* oW2  = (const float*)d_in[10];
  const float* ob2  = (const float*)d_in[11];

  float* agg = (float*)d_ws;   // [kM][kCOLS] f32, 26.4 MB (L2-resident)

  const int n4 = (kM * kCOLS) / 4;
  zero_f4<<<(n4 + 255) / 256, 256, 0, stream>>>((float4*)agg, n4);

  const int tilesB = kN / 16;      // 31250
  attn_scatter_kernel<<<(tilesB + 7) / 8, 256, kSmemB, stream>>>(
      feat, pts, ctr, lbl, aW1, ab1, aW2, ab2, agg, tilesB);

  const int tilesC = kM / 16;      // 3125
  out_mlp_kernel<<<(tilesC + 7) / 8, 256, kSmemC, stream>>>(
      agg, oW1, ob1, oW2, ob2, (float*)d_out, tilesC);
}
